// Attention_24524263260186
// MI455X (gfx1250) — matrix operand order repaired and checked by reading, not compiled, not run on hardware
//
#include <hip/hip_runtime.h>
#include <hip/hip_bf16.h>

typedef __attribute__((ext_vector_type(16))) _Float16 v16h;
typedef __attribute__((ext_vector_type(8)))  _Float16 v8h;
typedef __attribute__((ext_vector_type(8)))  float    v8f;

#define WMMA_F16(a, b, c) \
  __builtin_amdgcn_wmma_f32_16x16x32_f16(false, (a), false, (b), (short)0, (c), false, false)

// Async copy of 64 contiguous bytes global -> LDS (per lane), ASYNCcnt-tracked.
// LDS addr and mem addr both advance by INST_OFFSET per CDNA5 ISA 10.x async ops.
__device__ __forceinline__ void async_copy64(unsigned lds, const void* g) {
  asm volatile(
      "global_load_async_to_lds_b128 %0, %1, off\n\t"
      "global_load_async_to_lds_b128 %0, %1, off offset:16\n\t"
      "global_load_async_to_lds_b128 %0, %1, off offset:32\n\t"
      "global_load_async_to_lds_b128 %0, %1, off offset:48"
      :: "v"(lds), "v"(g) : "memory");
}
__device__ __forceinline__ void async_wait0() {
  asm volatile("s_wait_asynccnt 0" ::: "memory");
}

union h2pack { _Float16 h[2]; unsigned u; };

// ---------------------------------------------------------------------------
// Kernel 1: LayerNorm over last dim (1024), write normalized activations f16.
// ---------------------------------------------------------------------------
__global__ __launch_bounds__(256) void ln_kernel(
    const float* __restrict__ x, const float* __restrict__ gamma,
    const float* __restrict__ beta, _Float16* __restrict__ xn) {
  const int row = blockIdx.x;
  const int t = threadIdx.x;
  const float* xr = x + (size_t)row * 1024;

  float v[4];
  float s = 0.f;
#pragma unroll
  for (int i = 0; i < 4; ++i) { v[i] = xr[t + 256 * i]; s += v[i]; }

  __shared__ float red[256];
  red[t] = s;
  __syncthreads();
  for (int off = 128; off > 0; off >>= 1) {
    if (t < off) red[t] += red[t + off];
    __syncthreads();
  }
  const float mu = red[0] * (1.f / 1024.f);
  __syncthreads();

  float vs = 0.f;
#pragma unroll
  for (int i = 0; i < 4; ++i) { float d = v[i] - mu; vs += d * d; }
  red[t] = vs;
  __syncthreads();
  for (int off = 128; off > 0; off >>= 1) {
    if (t < off) red[t] += red[t + off];
    __syncthreads();
  }
  const float rstd = rsqrtf(red[0] * (1.f / 1024.f) + 1e-5f);

#pragma unroll
  for (int i = 0; i < 4; ++i) {
    int c = t + 256 * i;
    xn[(size_t)row * 1024 + c] = (_Float16)((v[i] - mu) * rstd * gamma[c] + beta[c]);
  }
}

// ---------------------------------------------------------------------------
// Kernel 2: QKV GEMM. C[4096,3072] = xn_f16[4096,1024] @ w_qkv_f32[1024,3072].
// Tile 128x64 (BK=32), 4 waves; each wave: 2 A-frags x 4 B-frags = 8 WMMAs/step.
// A tile staged via async global->LDS; W staged f32->f16 with packed b32 stores.
// ---------------------------------------------------------------------------
__global__ __launch_bounds__(128) void qkv_gemm(
    const _Float16* __restrict__ A, const float* __restrict__ W,
    _Float16* __restrict__ Q, _Float16* __restrict__ K, _Float16* __restrict__ V) {
  __shared__ __align__(16) _Float16 As[128][40];  // [m][k], pad 8
  __shared__ __align__(16) _Float16 Bs[64][40];   // transposed: [n][k], pad 8

  const int n0 = blockIdx.x * 64;
  const int m0 = blockIdx.y * 128;
  const int t = threadIdx.x;
  const int wave = t >> 5, lane = t & 31, l16 = lane & 15, h = lane >> 4;

  v8f c[2][4] = {};

  const unsigned ldsA = (unsigned)(size_t)&As[t][0];
  const int kp = t >> 3;            // k-pair 0..15
  const int csn = (t & 7) * 8;      // n seg

  for (int k0 = 0; k0 < 1024; k0 += 32) {
    // A tile (128x32 f16): 64B contiguous per thread, async to LDS
    async_copy64(ldsA, A + (size_t)(m0 + t) * 1024 + k0);

    // B tile (32x64 f32 -> f16, transposed [n][k]); pack k-pairs into b32
    {
      const float* s0 = W + (size_t)(k0 + 2 * kp) * 3072 + n0 + csn;
      const float* s1 = s0 + 3072;
#pragma unroll
      for (int j = 0; j < 8; ++j) {
        h2pack p;
        p.h[0] = (_Float16)s0[j];
        p.h[1] = (_Float16)s1[j];
        *(unsigned*)&Bs[csn + j][2 * kp] = p.u;
      }
    }
    if (k0 + 32 < 1024) {
      __builtin_prefetch(A + (size_t)(m0 + t) * 1024 + k0 + 32, 0, 1);
      __builtin_prefetch(W + (size_t)(k0 + 32 + (t >> 2)) * 3072 + n0 + (t & 3) * 16, 0, 1);
    }
    async_wait0();
    __syncthreads();

    v16h a[2];
#pragma unroll
    for (int fa = 0; fa < 2; ++fa) {
      const int arow = wave * 32 + fa * 16 + l16;
#pragma unroll
      for (int e = 0; e < 16; ++e) {
        int kk = e + 16 * h;
        a[fa][e] = As[arow][kk];
      }
    }
#pragma unroll
    for (int nc = 0; nc < 4; ++nc) {
      v16h b;
      const int bcol = nc * 16 + l16;
#pragma unroll
      for (int e = 0; e < 16; ++e) b[e] = Bs[bcol][16 * h + e];
      c[0][nc] = WMMA_F16(a[0], b, c[0][nc]);
      c[1][nc] = WMMA_F16(a[1], b, c[1][nc]);
    }
    __syncthreads();
  }

  // block-uniform destination routing
  const int part = n0 >> 10;                       // 0:Q 1:K 2:V
  _Float16* dst = (part == 0) ? Q : (part == 1) ? K : V;
  const float scale = (part == 0) ? 0.125f : 1.0f; // q * dim_head^-0.5
  const int ncol = n0 & 1023;

#pragma unroll
  for (int fa = 0; fa < 2; ++fa) {
#pragma unroll
    for (int nc = 0; nc < 4; ++nc) {
#pragma unroll
      for (int i = 0; i < 8; ++i) {
        int gm = m0 + wave * 32 + fa * 16 + i + 8 * h;
        int cc = ncol + nc * 16 + l16;
        int head = cc >> 6, d = cc & 63;
        int bb = gm >> 11, nrow = gm & 2047;
        dst[((size_t)(bb * 16 + head) * 2048 + nrow) * 64 + d] =
            (_Float16)(c[fa][nc][i] * scale);
      }
    }
  }
}

// ---------------------------------------------------------------------------
// Kernel 3: Flash attention. grid = (qblocks=32, b*h=32), block = 128 (4 waves).
// K tile staged async global->LDS; V transposed with packed b32 stores.
// ---------------------------------------------------------------------------
__global__ __launch_bounds__(128) void attn_kernel(
    const _Float16* __restrict__ Q, const _Float16* __restrict__ K,
    const _Float16* __restrict__ V, _Float16* __restrict__ O) {
  __shared__ __align__(16) _Float16 Ks[64][72];        // [key][d]
  __shared__ __align__(16) _Float16 Vs[64][72];        // transposed: [d][key]
  __shared__ __align__(16) _Float16 Ps[4][16][72];     // per-wave P tile [qrow][key]

  const int bh = blockIdx.y;
  const int q0 = blockIdx.x * 64;
  const _Float16* Qb = Q + (size_t)bh * 2048 * 64;
  const _Float16* Kb = K + (size_t)bh * 2048 * 64;
  const _Float16* Vb = V + (size_t)bh * 2048 * 64;

  const int t = threadIdx.x;
  const int wave = t >> 5, lane = t & 31, l16 = lane & 15, h = lane >> 4;

  // Q fragments for this wave's 16 rows (K-dim 64 = 2 fragments)
  v16h qa[2];
  const int qrow = q0 + wave * 16 + l16;
#pragma unroll
  for (int f = 0; f < 2; ++f)
#pragma unroll
    for (int e = 0; e < 16; ++e) {
      int d = 32 * f + 16 * h + e;
      qa[f][e] = Qb[(size_t)qrow * 64 + d];
    }

  v8f o[4] = {};
  float m_i[8], l_i[8];
#pragma unroll
  for (int i = 0; i < 8; ++i) { m_i[i] = -1e30f; l_i[i] = 0.f; }

  const int kr = t >> 1, kcs = (t & 1) * 32;       // K staging: 64B/thread
  const unsigned ldsK = (unsigned)(size_t)&Ks[kr][kcs];
  const int vp = t >> 2, vds = (t & 3) * 16;       // V staging: key-pair, d seg

  for (int j0 = 0; j0 < 2048; j0 += 64) {
    // K tile async copy (row-major, contiguous)
    async_copy64(ldsK, Kb + (size_t)(j0 + kr) * 64 + kcs);
    // V tile transposed: pack key-pairs along contiguous key dim
    {
      const _Float16* v0 = Vb + (size_t)(j0 + 2 * vp) * 64 + vds;
      const _Float16* v1 = v0 + 64;
#pragma unroll
      for (int j = 0; j < 16; ++j) {
        h2pack p;
        p.h[0] = v0[j];
        p.h[1] = v1[j];
        *(unsigned*)&Vs[vds + j][2 * vp] = p.u;
      }
    }
    if (j0 + 64 < 2048) {
      __builtin_prefetch(Kb + (size_t)(j0 + 64 + kr) * 64 + kcs, 0, 1);
      __builtin_prefetch(Vb + (size_t)(j0 + 64 + kr) * 64 + kcs, 0, 1);
    }
    async_wait0();
    __syncthreads();

    // S = Q @ K^T : four 16x16 tiles across 64 keys
    v8f s[4];
#pragma unroll
    for (int tt = 0; tt < 4; ++tt) {
      v16h b0, b1;
      const int key = tt * 16 + l16;
#pragma unroll
      for (int e = 0; e < 16; ++e) {
        b0[e] = Ks[key][16 * h + e];
        b1[e] = Ks[key][32 + 16 * h + e];
      }
      v8f z = {};
      z = WMMA_F16(qa[0], b0, z);
      s[tt] = WMMA_F16(qa[1], b1, z);
    }

    // online softmax (row stats striped per C-layout: row = i + 8h)
    float m_new[8], alpha[8];
#pragma unroll
    for (int i = 0; i < 8; ++i) {
      float rm = fmaxf(fmaxf(s[0][i], s[1][i]), fmaxf(s[2][i], s[3][i]));
#pragma unroll
      for (int mk = 8; mk >= 1; mk >>= 1) rm = fmaxf(rm, __shfl_xor(rm, mk, 32));
      m_new[i] = fmaxf(m_i[i], rm);
      alpha[i] = __expf(m_i[i] - m_new[i]);
      m_i[i] = m_new[i];
    }
#pragma unroll
    for (int tt = 0; tt < 4; ++tt)
#pragma unroll
      for (int i = 0; i < 8; ++i) s[tt][i] = __expf(s[tt][i] - m_new[i]);
#pragma unroll
    for (int i = 0; i < 8; ++i) {
      float rs = s[0][i] + s[1][i] + s[2][i] + s[3][i];
#pragma unroll
      for (int mk = 8; mk >= 1; mk >>= 1) rs += __shfl_xor(rs, mk, 32);
      l_i[i] = l_i[i] * alpha[i] + rs;
    }
#pragma unroll
    for (int tt = 0; tt < 4; ++tt)
#pragma unroll
      for (int i = 0; i < 8; ++i) o[tt][i] *= alpha[i];

    // park P in LDS (C-layout -> A-layout shuffle through memory)
#pragma unroll
    for (int tt = 0; tt < 4; ++tt)
#pragma unroll
      for (int i = 0; i < 8; ++i)
        Ps[wave][i + 8 * h][tt * 16 + l16] = (_Float16)s[tt][i];
    asm volatile("s_wait_dscnt 0" ::: "memory");  // wave-local LDS RAW fence

    // O += P @ V
    v16h pa[2];
#pragma unroll
    for (int f = 0; f < 2; ++f)
#pragma unroll
      for (int e = 0; e < 16; ++e) {
        int kk = 32 * f + 16 * h + e;
        pa[f][e] = Ps[wave][l16][kk];
      }
#pragma unroll
    for (int tt = 0; tt < 4; ++tt) {
      v16h vb0, vb1;
      const int col = tt * 16 + l16;
#pragma unroll
      for (int e = 0; e < 16; ++e) {
        vb0[e] = Vs[col][16 * h + e];
        vb1[e] = Vs[col][32 + 16 * h + e];
      }
      o[tt] = WMMA_F16(pa[0], vb0, o[tt]);
      o[tt] = WMMA_F16(pa[1], vb1, o[tt]);
    }
    __syncthreads();
  }

  // normalize and store in [b, n, h*64] layout
  const int head = bh & 15, bb = bh >> 4;
#pragma unroll
  for (int tt = 0; tt < 4; ++tt)
#pragma unroll
    for (int i = 0; i < 8; ++i) {
      int row = q0 + wave * 16 + i + 8 * h;
      int col = head * 64 + tt * 16 + l16;
      float val = o[tt][i] / l_i[i];
      O[((size_t)bb * 2048 + row) * 1024 + col] = (_Float16)val;
    }
}

// ---------------------------------------------------------------------------
// Kernel 4: Output projection. out[4096,1024] = attn_f16 @ w_out_f32 + b_out.
// ---------------------------------------------------------------------------
__global__ __launch_bounds__(128) void out_gemm(
    const _Float16* __restrict__ A, const float* __restrict__ W,
    const float* __restrict__ bias, float* __restrict__ out) {
  __shared__ __align__(16) _Float16 As[128][40];
  __shared__ __align__(16) _Float16 Bs[64][40];

  const int n0 = blockIdx.x * 64;
  const int m0 = blockIdx.y * 128;
  const int t = threadIdx.x;
  const int wave = t >> 5, lane = t & 31, l16 = lane & 15, h = lane >> 4;

  v8f c[2][4] = {};

  const unsigned ldsA = (unsigned)(size_t)&As[t][0];
  const int kp = t >> 3;
  const int csn = (t & 7) * 8;

  for (int k0 = 0; k0 < 1024; k0 += 32) {
    async_copy64(ldsA, A + (size_t)(m0 + t) * 1024 + k0);
    {
      const float* s0 = W + (size_t)(k0 + 2 * kp) * 1024 + n0 + csn;
      const float* s1 = s0 + 1024;
#pragma unroll
      for (int j = 0; j < 8; ++j) {
        h2pack p;
        p.h[0] = (_Float16)s0[j];
        p.h[1] = (_Float16)s1[j];
        *(unsigned*)&Bs[csn + j][2 * kp] = p.u;
      }
    }
    if (k0 + 32 < 1024) {
      __builtin_prefetch(A + (size_t)(m0 + t) * 1024 + k0 + 32, 0, 1);
      __builtin_prefetch(W + (size_t)(k0 + 32 + (t >> 2)) * 1024 + n0 + (t & 3) * 16, 0, 1);
    }
    async_wait0();
    __syncthreads();

    v16h a[2];
#pragma unroll
    for (int fa = 0; fa < 2; ++fa) {
      const int arow = wave * 32 + fa * 16 + l16;
#pragma unroll
      for (int e = 0; e < 16; ++e) {
        int kk = e + 16 * h;
        a[fa][e] = As[arow][kk];
      }
    }
#pragma unroll
    for (int nc = 0; nc < 4; ++nc) {
      v16h b;
      const int bcol = nc * 16 + l16;
#pragma unroll
      for (int e = 0; e < 16; ++e) b[e] = Bs[bcol][16 * h + e];
      c[0][nc] = WMMA_F16(a[0], b, c[0][nc]);
      c[1][nc] = WMMA_F16(a[1], b, c[1][nc]);
    }
    __syncthreads();
  }

#pragma unroll
  for (int fa = 0; fa < 2; ++fa) {
#pragma unroll
    for (int nc = 0; nc < 4; ++nc) {
#pragma unroll
      for (int i = 0; i < 8; ++i) {
        int gm = m0 + wave * 32 + fa * 16 + i + 8 * h;
        int gn = n0 + nc * 16 + l16;
        out[(size_t)gm * 1024 + gn] = c[fa][nc][i] + bias[gn];
      }
    }
  }
}

// ---------------------------------------------------------------------------
// Host launcher. Workspace: xn 8MB | Q 8MB | K 8MB | V 8MB | attn_out 8MB.
// ---------------------------------------------------------------------------
extern "C" void kernel_launch(void* const* d_in, const int* in_sizes, int n_in,
                              void* d_out, int out_size, void* d_ws, size_t ws_size,
                              hipStream_t stream) {
  (void)in_sizes; (void)n_in; (void)out_size; (void)ws_size;

  const float* x     = (const float*)d_in[0];
  const float* gamma = (const float*)d_in[1];
  const float* beta  = (const float*)d_in[2];
  const float* w_qkv = (const float*)d_in[3];
  const float* w_out = (const float*)d_in[4];
  const float* b_out = (const float*)d_in[5];
  float* out = (float*)d_out;

  char* ws = (char*)d_ws;
  const size_t MB8 = (size_t)8 * 1024 * 1024;
  _Float16* xn = (_Float16*)(ws);
  _Float16* q  = (_Float16*)(ws + 1 * MB8);
  _Float16* k  = (_Float16*)(ws + 2 * MB8);
  _Float16* v  = (_Float16*)(ws + 3 * MB8);
  _Float16* ao = (_Float16*)(ws + 4 * MB8);

  ln_kernel<<<dim3(4096), dim3(256), 0, stream>>>(x, gamma, beta, xn);
  qkv_gemm<<<dim3(48, 32), dim3(128), 0, stream>>>(xn, w_qkv, q, k, v);
  attn_kernel<<<dim3(32, 32), dim3(128), 0, stream>>>(q, k, v, ao);
  out_gemm<<<dim3(16, 32), dim3(128), 0, stream>>>(ao, w_out, b_out, out);
}
